// MultiHeadSelfAttentionGating_20469814133552
// MI455X (gfx1250) — compile-verified, matrix operand order checked by reading, not run on hardware
//
#include <hip/hip_runtime.h>
#include <hip/hip_bf16.h>
#include <math.h>

// ---------------------------------------------------------------------------
// MI455X (gfx1250) fused MHA+gating.
// - One-time fp32->bf16 conversion of activations/weights
// - bf16 WMMA (fp32 accum) for all GEMMs and attention matmuls
// - GLOBAL_LOAD_ASYNC_TO_LDS_B128 staging, double-buffered LDS pipeline with
//   peeled epilogue: steady state issues tile i+1, waits asynccnt<=1, computes
//   tile i while i+1 is in flight.
// - wave32; 256-thread blocks = 8 waves; each wave owns two 16x16 WMMA tiles
// ---------------------------------------------------------------------------

typedef __attribute__((ext_vector_type(16))) __bf16 v16bf;
typedef __attribute__((ext_vector_type(2)))  __bf16 v2bf;
typedef __attribute__((ext_vector_type(8)))  float  v8f;

struct alignas(16) U128 { unsigned x, y, z, w; };
struct alignas(16) F128 { float x, y, z, w; };

union Frag  { U128 u[2]; v16bf v; __bf16 e[16]; };
union Pack8 { U128 u; __bf16 e[8]; };

// Pack two fp32 -> packed bf16 pair (one dword).
static __device__ __forceinline__ unsigned pk2(float a, float b) {
#if __has_builtin(__builtin_amdgcn_cvt_pk_bf16_f32)
  union { v2bf v; unsigned u; } c;
  c.v = __builtin_amdgcn_cvt_pk_bf16_f32(a, b);
  return c.u;
#else
  union { __bf16 e[2]; unsigned u; } c;
  c.e[0] = (__bf16)a; c.e[1] = (__bf16)b;
  return c.u;
#endif
}

// Per-lane async copy of 16B global -> LDS (gfx1250, tracked by ASYNCcnt).
static __device__ __forceinline__ void async_cp16(const void* gptr, void* lptr) {
  const unsigned ldsoff = (unsigned)(size_t)lptr;   // LDS aperture: low 32b = offset
  asm volatile("global_load_async_to_lds_b128 %0, %1, off"
               :: "v"(ldsoff), "v"(gptr) : "memory");
}
static __device__ __forceinline__ void async_wait0() {
  asm volatile("s_wait_asynccnt 0x0" ::: "memory");
}
static __device__ __forceinline__ void async_wait1() {
  asm volatile("s_wait_asynccnt 0x1" ::: "memory");
}
static __device__ __forceinline__ void async_wait2() {
  asm volatile("s_wait_asynccnt 0x2" ::: "memory");
}

// Load one 16-rows x 32-k bf16 fragment from LDS in CDNA5 WMMA A/B layout:
// lane<16: row=lane,    K = {0..7, 16..23}
// lane>=16: row=lane-16, K = {8..15, 24..31}
static __device__ __forceinline__ v16bf load_frag_lds(const __bf16* base, int stride, int lane) {
  const int lr = lane & 15, hi = lane >> 4;
  const __bf16* p = base + lr * stride + hi * 8;
  Frag f;
  f.u[0] = *(const U128*)(p);
  f.u[1] = *(const U128*)(p + 16);
  return f.v;
}

// ---------------------------------------------------------------------------
// fp32 -> bf16 bulk convert: 8 elements/thread, 128-bit in / 128-bit out.
// ---------------------------------------------------------------------------
__global__ __launch_bounds__(256)
void k_cvt_bf16(const float* __restrict__ src, __bf16* __restrict__ dst) {
  const size_t i = ((size_t)blockIdx.x * 256 + threadIdx.x) * 8;
  F128 a0 = *(const F128*)(src + i);
  F128 a1 = *(const F128*)(src + i + 4);
  U128 o;
  o.x = pk2(a0.x, a0.y); o.y = pk2(a0.z, a0.w);
  o.z = pk2(a1.x, a1.y); o.w = pk2(a1.z, a1.w);
  *(U128*)(dst + i) = o;
}

// ---------------------------------------------------------------------------
// GEMM: C[M,N] (fp32, +bias) = A[M,K] x B[K,N], A/B bf16.
// 64x64 block tile, double-buffered LDS pipeline with peeled epilogue:
//   A via async-to-LDS b128; B held in regs one tile ahead, transposed into
//   LDS [n][k] at the top of each iteration.
// ---------------------------------------------------------------------------
__global__ __launch_bounds__(256)
void k_gemm_bf16_wmma(const __bf16* __restrict__ A, const __bf16* __restrict__ Bm,
                      const float* __restrict__ bias, float* __restrict__ C,
                      int M, int N, int K) {
  __shared__ __bf16 sA[2][64 * 32];       // [row][k]
  __shared__ __bf16 sB[2][64 * 32];       // [n][k]

  const int tid  = threadIdx.x;
  const int lane = tid & 31;
  const int wave = tid >> 5;
  const int m0 = blockIdx.y * 64;
  const int n0 = blockIdx.x * 64;
  const int rs = wave >> 1;
  const int cs = (wave & 1) * 2;

  const int arow = tid >> 2, aseg = (tid & 3) * 8;   // A stage: 64 rows x 32 k
  const int bk   = tid >> 3, bseg = (tid & 7) * 8;   // B stage: 32 k  x 64 n

  const __bf16* pa = A + (size_t)(m0 + arow) * K + aseg;
  const __bf16* pb = Bm + (size_t)bk * N + n0 + bseg;
  const size_t bstep = (size_t)32 * N;

  // pipeline prologue: tile 0 in flight
  async_cp16(pa, &sA[0][arow * 32 + aseg]);
  Pack8 breg; breg.u = *(const U128*)pb;

  v8f acc0 = {}; v8f acc1 = {};
  const int nIter = K >> 5;

  auto stageB = [&](int cur) {
#pragma unroll
    for (int e = 0; e < 8; ++e) sB[cur][(bseg + e) * 32 + bk] = breg.e[e];
  };
  auto compute = [&](int cur) {
    v16bf fa  = load_frag_lds(&sA[cur][(rs * 16) * 32], 32, lane);
    v16bf fb0 = load_frag_lds(&sB[cur][(cs * 16) * 32], 32, lane);
    v16bf fb1 = load_frag_lds(&sB[cur][((cs + 1) * 16) * 32], 32, lane);
    acc0 = __builtin_amdgcn_wmma_f32_16x16x32_bf16(false, fa, false, fb0, (short)0, acc0, false, false);
    acc1 = __builtin_amdgcn_wmma_f32_16x16x32_bf16(false, fa, false, fb1, (short)0, acc1, false, false);
  };

  // steady state: branch-free, always one tile in flight
  for (int i = 0; i < nIter - 1; ++i) {
    const int cur = i & 1;
    stageB(cur);                          // B(i) regs -> LDS
    pa += 32; pb += bstep;
    async_cp16(pa, &sA[cur ^ 1][arow * 32 + aseg]);   // A(i+1)
    breg.u = *(const U128*)pb;                        // B(i+1)
    __builtin_prefetch(pb + bstep, 0, 1);             // B(i+2) into L2
    async_wait1();                        // A(i) done; A(i+1) still in flight
    __syncthreads();                      // tile i fully staged
    compute(cur);
    __syncthreads();                      // readers done; buffers reusable
  }
  // peeled epilogue: last tile
  {
    const int cur = (nIter - 1) & 1;
    stageB(cur);
    async_wait0();
    __syncthreads();
    compute(cur);
  }

  const int lr = lane & 15, hi = lane >> 4;
  const int col0 = n0 + cs * 16 + lr;
  const int col1 = col0 + 16;
  const float bv0 = bias ? bias[col0] : 0.f;
  const float bv1 = bias ? bias[col1] : 0.f;
#pragma unroll
  for (int r = 0; r < 8; ++r) {
    const int row = m0 + rs * 16 + hi * 8 + r;
    C[(size_t)row * N + col0] = acc0[r] + bv0;
    C[(size_t)row * N + col1] = acc1[r] + bv1;
  }
}

// ---------------------------------------------------------------------------
// RoPE on q,k + pack q/k/v fp32 -> bf16 in [b,h,s,d] layout.
// ---------------------------------------------------------------------------
__global__ __launch_bounds__(256)
void k_rope_pack(const float* __restrict__ qkv, __bf16* __restrict__ qb,
                 __bf16* __restrict__ kb, __bf16* __restrict__ vb) {
  const int S = 2048, E = 1024, H = 16, D = 64;
  const int idx = blockIdx.x * 256 + threadIdx.x;     // b(1) s(11) h(4) i(5)
  const int i = idx & 31;
  const int h = (idx >> 5) & 15;
  const int s = (idx >> 9) & 2047;
  const int b = idx >> 20;

  const size_t row = ((size_t)b * S + s) * (3 * E);
  const float* q = qkv + row + 0 * E + h * D;
  const float* k = qkv + row + 1 * E + h * D;
  const float* v = qkv + row + 2 * E + h * D;

  // inv_freq = theta^(-2i/D); ln(10000)=9.21034037...
  const float ang = (float)s * __expf(-(float)(2 * i) * (9.2103403719761836f / 64.f));
  const float c = __cosf(ang), sn = __sinf(ang);

  const size_t ob = (((size_t)b * H + h) * S + s) * D;
  const float qe = q[2 * i], qo = q[2 * i + 1];
  qb[ob + i]      = (__bf16)(qe * c - qo * sn);
  qb[ob + 32 + i] = (__bf16)(qe * sn + qo * c);
  const float ke = k[2 * i], ko = k[2 * i + 1];
  kb[ob + i]      = (__bf16)(ke * c - ko * sn);
  kb[ob + 32 + i] = (__bf16)(ke * sn + ko * c);
  *(unsigned*)&vb[ob + 2 * i] = pk2(v[2 * i], v[2 * i + 1]);
}

// ---------------------------------------------------------------------------
// Flash attention: block = (b,h, 64 q rows); double-buffered 64-row K/V tiles
// with peeled epilogue. K via async-to-LDS; V reg-held one tile ahead,
// transposed into LDS [d][kv] so P.V B-fragments are ds_load_b128.
// Online softmax fp32 -> packed-bf16 P.
// ---------------------------------------------------------------------------
__global__ __launch_bounds__(256)
void k_flash_attn(const __bf16* __restrict__ qb, const __bf16* __restrict__ kb,
                  const __bf16* __restrict__ vb, float* __restrict__ outp) {
  const int S = 2048, D = 64, E = 1024;
  const float scale = 0.125f;                         // 1/sqrt(64)

  __shared__ __bf16 sQ[64 * 64];                      // [q][d]
  __shared__ __bf16 sK2[2][64 * 64];                  // [kv][d]
  __shared__ __bf16 sVt[2][64 * 64];                  // [d][kv] (transposed)
  __shared__ float  sS[64 * 64];                      // fp32 scores
  __shared__ __bf16 sP[64 * 64];                      // bf16 probabilities
  __shared__ float  sM[64], sL[64], sAl[64];

  const int tid = threadIdx.x, lane = tid & 31, wave = tid >> 5;
  const int q0 = blockIdx.x * 64;
  const int bh = blockIdx.y;
  const int b = bh >> 4, h = bh & 15;
  const size_t hb = (size_t)bh * S * D;

  // per-thread staging coords: two 8-element (16B) segments of the 64x64 tile
  const int u0 = tid * 2;
  const int r0 = u0 >> 3, c80 = (u0 & 7) * 8;
  const int r1 = (u0 + 1) >> 3, c81 = ((u0 + 1) & 7) * 8;

  // prologue: Q (async), K tile 0 (async), V tile 0 (regs)
  async_cp16(&qb[hb + (size_t)(q0 + r0) * D + c80], &sQ[r0 * 64 + c80]);
  async_cp16(&qb[hb + (size_t)(q0 + r1) * D + c81], &sQ[r1 * 64 + c81]);
  const __bf16* pk0 = &kb[hb + (size_t)r0 * D + c80];
  const __bf16* pk1 = &kb[hb + (size_t)r1 * D + c81];
  const __bf16* pv0 = &vb[hb + (size_t)r0 * D + c80];
  const __bf16* pv1 = &vb[hb + (size_t)r1 * D + c81];
  async_cp16(pk0, &sK2[0][r0 * 64 + c80]);
  async_cp16(pk1, &sK2[0][r1 * 64 + c81]);
  Pack8 v0, v1;
  v0.u = *(const U128*)pv0;
  v1.u = *(const U128*)pv1;

  if (tid < 64) { sM[tid] = -1e30f; sL[tid] = 0.f; }

  const int qs = wave >> 1;
  const int cs = (wave & 1) * 2;                      // kv-sub (scores) / d-sub (output)
  const int lr = lane & 15, hi = lane >> 4;
  v8f o0 = {}, o1 = {};

  auto stageV = [&](int cur) {                        // V(i) regs -> LDS [d][kv]
#pragma unroll
    for (int e = 0; e < 8; ++e) sVt[cur][(c80 + e) * 64 + r0] = v0.e[e];
#pragma unroll
    for (int e = 0; e < 8; ++e) sVt[cur][(c81 + e) * 64 + r1] = v1.e[e];
  };

  auto tile = [&](int cur) {
    // ---- scores: S = Q K^T * scale ----
#pragma unroll
    for (int j = 0; j < 2; ++j) {
      const int ks = cs + j;
      v8f sc = {};
#pragma unroll
      for (int d0 = 0; d0 < 64; d0 += 32) {
        v16bf fq = load_frag_lds(&sQ[(qs * 16) * 64 + d0], 64, lane);
        v16bf fk = load_frag_lds(&sK2[cur][(ks * 16) * 64 + d0], 64, lane);
        sc = __builtin_amdgcn_wmma_f32_16x16x32_bf16(false, fq, false, fk, (short)0, sc, false, false);
      }
#pragma unroll
      for (int r = 0; r < 8; ++r)
        sS[(qs * 16 + hi * 8 + r) * 64 + ks * 16 + lr] = sc[r] * scale;
    }
    __syncthreads();

    // ---- online softmax: one thread per q row; emit packed bf16 P ----
    if (tid < 64) {
      const float mold = sM[tid];
      float mnew = mold;
      const float* rowp = &sS[tid * 64];
#pragma unroll 8
      for (int c = 0; c < 64; ++c) mnew = fmaxf(mnew, rowp[c]);
      const float alpha = __expf(mold - mnew);
      float l = sL[tid] * alpha;
      unsigned* prow = (unsigned*)&sP[tid * 64];
#pragma unroll 8
      for (int c = 0; c < 64; c += 2) {
        const float p0 = __expf(rowp[c] - mnew);
        const float p1 = __expf(rowp[c + 1] - mnew);
        prow[c >> 1] = pk2(p0, p1);
        l += p0 + p1;
      }
      sM[tid] = mnew; sL[tid] = l; sAl[tid] = alpha;
    }
    __syncthreads();

    // ---- rescale accumulators, then O += P V ----
#pragma unroll
    for (int r = 0; r < 8; ++r) {
      const float a = sAl[qs * 16 + hi * 8 + r];
      o0[r] *= a; o1[r] *= a;
    }
#pragma unroll
    for (int kb2 = 0; kb2 < 64; kb2 += 32) {
      v16bf fp  = load_frag_lds(&sP[(qs * 16) * 64 + kb2], 64, lane);
      v16bf fv0 = load_frag_lds(&sVt[cur][(cs * 16) * 64 + kb2], 64, lane);
      v16bf fv1 = load_frag_lds(&sVt[cur][((cs + 1) * 16) * 64 + kb2], 64, lane);
      o0 = __builtin_amdgcn_wmma_f32_16x16x32_bf16(false, fp, false, fv0, (short)0, o0, false, false);
      o1 = __builtin_amdgcn_wmma_f32_16x16x32_bf16(false, fp, false, fv1, (short)0, o1, false, false);
    }
  };

  const int nT = S / 64;                              // 32 kv tiles
  // steady state: branch-free, K(i+1)/V(i+1) always in flight
  for (int i = 0; i < nT - 1; ++i) {
    const int cur = i & 1;
    __syncthreads();                                  // P.V(i-1) reads complete
    stageV(cur);
    pk0 += 4096; pk1 += 4096; pv0 += 4096; pv1 += 4096;
    async_cp16(pk0, &sK2[cur ^ 1][r0 * 64 + c80]);    // K(i+1)
    async_cp16(pk1, &sK2[cur ^ 1][r1 * 64 + c81]);
    v0.u = *(const U128*)pv0;                         // V(i+1)
    v1.u = *(const U128*)pv1;
    async_wait2();                                    // K(i) done (Q done too)
    __syncthreads();                                  // tile i fully staged
    tile(cur);
  }
  // peeled epilogue: last kv tile
  {
    const int cur = (nT - 1) & 1;
    __syncthreads();
    stageV(cur);
    async_wait0();
    __syncthreads();
    tile(cur);
  }

  // ---- normalize and store [b,s,h,d] ----
#pragma unroll
  for (int r = 0; r < 8; ++r) {
    const int row = qs * 16 + hi * 8 + r;
    const float inv = 1.f / sL[row];
    const size_t orow = ((size_t)b * S + q0 + row) * E + h * D;
    outp[orow + cs * 16 + lr]       = o0[r] * inv;
    outp[orow + (cs + 1) * 16 + lr] = o1[r] * inv;
  }
}

// ---------------------------------------------------------------------------
// gated(bf16) = attn * sigmoid(gate_pre); 2 elems/thread, packed dword store.
// ---------------------------------------------------------------------------
__global__ __launch_bounds__(256)
void k_gate_mul(const float* __restrict__ attn, const float* __restrict__ gpre,
                __bf16* __restrict__ gated) {
  const size_t i = ((size_t)blockIdx.x * 256 + threadIdx.x) * 2;
  const float g0 = 1.f / (1.f + __expf(-gpre[i]));
  const float g1 = 1.f / (1.f + __expf(-gpre[i + 1]));
  *(unsigned*)&gated[i] = pk2(attn[i] * g0, attn[i + 1] * g1);
}

// ---------------------------------------------------------------------------
extern "C" void kernel_launch(void* const* d_in, const int* in_sizes, int n_in,
                              void* d_out, int out_size, void* d_ws, size_t ws_size,
                              hipStream_t stream) {
  (void)in_sizes; (void)n_in; (void)out_size; (void)ws_size;
  const float* x      = (const float*)d_in[0];
  const float* w_qkv  = (const float*)d_in[1];
  const float* w_gate = (const float*)d_in[2];
  const float* w_out  = (const float*)d_in[3];
  const float* b_out  = (const float*)d_in[4];

  const int Bd = 2, S = 2048, E = 1024, H = 16;
  const int M = Bd * S;                                  // 4096

  char* ws = (char*)d_ws;
  // layout (bytes):
  //   [0, 50.3M)      qkv fp32 (dead after rope) -> reused for attn & gated
  //   [50.3M, 67.1M)  gate_pre fp32
  //   [67.1M, 92.3M)  q/k/v bf16 [b,h,s,d]
  //   [92.3M, 111.1M) bf16 copies of x, w_qkv, w_gate, w_out
  float*  qkv   = (float*)(ws);
  float*  attn  = (float*)(ws);                          // alias (qkv dead)
  __bf16* gated = (__bf16*)(ws + 16777216);              // alias (disjoint)
  float*  gpre  = (float*)(ws + 50331648);
  __bf16* qb    = (__bf16*)(ws + 67108864);
  __bf16* kbuf  = (__bf16*)(ws + 67108864 + 8388608);
  __bf16* vbuf  = (__bf16*)(ws + 67108864 + 16777216);
  __bf16* xb    = (__bf16*)(ws + 92274688);              // 4,194,304 elems
  __bf16* wqb   = (__bf16*)(ws + 100663296);             // 3,145,728 elems
  __bf16* wgb   = (__bf16*)(ws + 106954752);             // 1,048,576 elems
  __bf16* wob   = (__bf16*)(ws + 109051904);             // 1,048,576 elems

  dim3 blk(256);
  k_cvt_bf16<<<dim3(2048), blk, 0, stream>>>(x, xb);
  k_cvt_bf16<<<dim3(1536), blk, 0, stream>>>(w_qkv, wqb);
  k_cvt_bf16<<<dim3(512),  blk, 0, stream>>>(w_gate, wgb);
  k_cvt_bf16<<<dim3(512),  blk, 0, stream>>>(w_out, wob);

  k_gemm_bf16_wmma<<<dim3(3 * E / 64, M / 64), blk, 0, stream>>>(xb, wqb, nullptr, qkv, M, 3 * E, E);
  k_gemm_bf16_wmma<<<dim3(E / 64, M / 64), blk, 0, stream>>>(xb, wgb, nullptr, gpre, M, E, E);
  k_rope_pack<<<dim3((Bd * S * H * 32) / 256), blk, 0, stream>>>(qkv, qb, kbuf, vbuf);
  k_flash_attn<<<dim3(S / 64, Bd * H), blk, 0, stream>>>(qb, kbuf, vbuf, attn);
  k_gate_mul<<<dim3(M * E / 512), blk, 0, stream>>>(attn, gpre, gated);
  k_gemm_bf16_wmma<<<dim3(E / 64, M / 64), blk, 0, stream>>>(gated, wob, b_out, (float*)d_out, M, E, E);
}
